// Model_146028888603
// MI455X (gfx1250) — matrix operand order repaired and checked by reading, not compiled, not run on hardware
//
#include <hip/hip_runtime.h>
#include <hip/hip_bf16.h>

// MI455X / gfx1250, wave32. bf16 WMMA (16x16x32) with f32 accumulate.

typedef __attribute__((ext_vector_type(16))) __bf16        v16bf;
typedef __attribute__((ext_vector_type(2)))  __bf16        v2bf;
typedef __attribute__((ext_vector_type(8)))  float         v8f;
typedef __attribute__((ext_vector_type(4)))  float         v4f;
typedef __attribute__((ext_vector_type(4)))  unsigned int  v4u;

#define LOG_2PI_F 1.8378770664093453f
#define EPS_F 1e-20f
#define LSTR 264   // padded LDS row stride (bf16 elems); 528 B = 33*16 -> 16B aligned rows

union Frag { v16bf v; unsigned int u[8]; v4u q[2]; };

// Hardware f32->bf16 (RNE).
__device__ __forceinline__ unsigned short f2bf(float f) {
    __bf16 b = (__bf16)f;
    return __builtin_bit_cast(unsigned short, b);
}
#if __has_builtin(__builtin_amdgcn_cvt_pk_bf16_f32)
// Single v_cvt_pk_bf16_f32: both halves in one VALU op.
__device__ __forceinline__ unsigned int pack2bf(float lo, float hi) {
    v2bf p = __builtin_amdgcn_cvt_pk_bf16_f32(lo, hi);
    return __builtin_bit_cast(unsigned int, p);
}
#else
__device__ __forceinline__ unsigned int pack2bf(float lo, float hi) {
    return (unsigned int)f2bf(lo) | ((unsigned int)f2bf(hi) << 16);
}
#endif
__device__ __forceinline__ float relu(float v) { return v > 0.0f ? v : 0.0f; }

// ---------------------------------------------------------------- init ------
__global__ void k_init(const float* __restrict__ We, const float* __restrict__ W1,
                       const float* __restrict__ W2, const float* __restrict__ W3,
                       unsigned short* __restrict__ We_bf, unsigned short* __restrict__ W1_bf,
                       unsigned short* __restrict__ W2_bf, unsigned short* __restrict__ W3_bf,
                       float* __restrict__ stats, float* __restrict__ kl_slot) {
    int tid = blockIdx.x * blockDim.x + threadIdx.x;
    int stride = gridDim.x * blockDim.x;
    for (int i = tid; i < 128 * 640; i += stride) We_bf[i] = f2bf(We[i]);
    for (int i = tid; i < 256 * 256; i += stride) {
        W1_bf[i] = f2bf(W1[i]); W2_bf[i] = f2bf(W2[i]); W3_bf[i] = f2bf(W3[i]);
    }
    for (int i = tid; i < 64 * 5 * 256; i += stride) stats[i] = 0.0f;
    if (tid == 0) *kl_slot = 0.0f;
}

// B-matrix 32x16 bf16 fragment: lane ln = N, group g selects K runs; two
// contiguous 8-element runs at col offsets g*8 and 16+g*8 = two b128 loads.
__device__ __forceinline__ Frag load_bfrag(const unsigned short* __restrict__ W,
                                           int rowstride, int o, int kb, int g) {
    Frag b;
    const unsigned short* p = W + (size_t)o * rowstride + kb + g * 8;
    b.q[0] = *(const v4u*)(p);
    b.q[1] = *(const v4u*)(p + 16);
    return b;
}

// A-matrix 16x32 bf16 fragment from a bf16 row-major buffer (global or LDS):
// two contiguous 8-element runs at col offsets g*8 and 16+g*8.
__device__ __forceinline__ Frag load_afrag_bf(const unsigned short* __restrict__ buf,
                                              int rowstride, int row, int kb, int g) {
    Frag a;
    const unsigned short* p = buf + (size_t)row * rowstride + kb + g * 8;
    a.q[0] = *(const v4u*)(p);
    a.q[1] = *(const v4u*)(p + 16);
    return a;
}

// ------------------------------------------------------------- encoder ------
// out[row,h] = sum_e x[row,e] * W_enc[h,e];  rows = 5120, e = 640, h = 128
// grid 320 x 256 threads (8 waves); wave w -> h col-tile w.
__global__ void k_encoder(const float* __restrict__ x,
                          const unsigned short* __restrict__ We_bf,
                          unsigned short* __restrict__ out_bf) {
    const int lane = threadIdx.x & 31;
    const int wave = threadIdx.x >> 5;
    const int g = lane >> 4, ln = lane & 15;
    const int rbase = blockIdx.x * 16;
    const int ob = wave * 16;
    const int arow = rbase + ln;
    v8f acc = {};
    for (int kb = 0; kb < 640; kb += 32) {
        Frag a;
        // A elements: cols [kb+g*8, +8) and [kb+16+g*8, +8): two float4 pairs, cvt+pack
        const float* p0 = x + (size_t)arow * 640 + kb + g * 8;
        v4f f0 = *(const v4f*)(p0);
        v4f f1 = *(const v4f*)(p0 + 4);
        v4f f2 = *(const v4f*)(p0 + 16);
        v4f f3 = *(const v4f*)(p0 + 20);
        a.u[0] = pack2bf(f0[0], f0[1]); a.u[1] = pack2bf(f0[2], f0[3]);
        a.u[2] = pack2bf(f1[0], f1[1]); a.u[3] = pack2bf(f1[2], f1[3]);
        a.u[4] = pack2bf(f2[0], f2[1]); a.u[5] = pack2bf(f2[2], f2[3]);
        a.u[6] = pack2bf(f3[0], f3[1]); a.u[7] = pack2bf(f3[2], f3[3]);
        Frag b = load_bfrag(We_bf, 640, ob + ln, kb, g);
        acc = __builtin_amdgcn_wmma_f32_16x16x32_bf16(false, a.v, false, b.v,
                                                      (short)0, acc, false, false);
    }
#pragma unroll
    for (int r = 0; r < 8; ++r) {
        int M = g * 8 + r;
        out_bf[(size_t)(rbase + M) * 128 + ob + ln] = f2bf(acc[r]);
    }
}

// -------------------------------------------------- factorized layer 1 ------
// A1[row,o] = sum_{d<128} out[row,d]*W1[o,d] ; A2 uses W1[o,128+d].
// grid (320, 2) x 512 threads (16 waves); wave w -> o col-tile w.
__global__ void k_layer1(const unsigned short* __restrict__ out_bf,
                         const unsigned short* __restrict__ W1_bf,
                         float* __restrict__ A1, float* __restrict__ A2) {
    const int lane = threadIdx.x & 31;
    const int wave = threadIdx.x >> 5;
    const int g = lane >> 4, ln = lane & 15;
    const int rbase = blockIdx.x * 16;
    const int ob = wave * 16;
    const int half = blockIdx.y;
    float* outp = half ? A2 : A1;
    const int arow = rbase + ln;
    v8f acc = {};
    for (int kb = 0; kb < 128; kb += 32) {
        Frag a = load_afrag_bf(out_bf, 128, arow, kb, g);
        Frag b = load_bfrag(W1_bf, 256, ob + ln, half * 128 + kb, g);
        acc = __builtin_amdgcn_wmma_f32_16x16x32_bf16(false, a.v, false, b.v,
                                                      (short)0, acc, false, false);
    }
#pragma unroll
    for (int r = 0; r < 8; ++r) {
        int M = g * 8 + r;
        outp[(size_t)(rbase + M) * 256 + ob + ln] = acc[r];
    }
}

// -------------------------------------------------- pairwise MLP core -------
// One block per (b,m,n). Wave w owns rows (j=w, k=0..15): a 16x256 stripe.
// h1 = relu(A1[b,m,j] + A2[b,n,k]) -> LDS(bf16); two WMMA layers (W2, W3),
// ping-pong LDS stripes; final relu-column-sum atomically accumulated into
// stats[b,m,:]. Dynamic LDS = 2 * 16 waves * 16 rows * LSTR * 2B = 270336 B.
__global__ void k_pairmlp(const float* __restrict__ A1, const float* __restrict__ A2,
                          const unsigned short* __restrict__ W2_bf,
                          const unsigned short* __restrict__ W3_bf,
                          float* __restrict__ stats) {
    extern __shared__ unsigned short lds[];
    const int lane = threadIdx.x & 31;
    const int wave = threadIdx.x >> 5;
    const int g = lane >> 4, ln = lane & 15;
    const int bid = blockIdx.x;
    const int b = bid / 25, rem = bid % 25;
    const int m = rem / 5, n = rem % 5;
    const int j = wave;

    unsigned short* src = lds + (size_t)wave * 16 * LSTR;
    unsigned short* dst = lds + (size_t)(16 + wave) * 16 * LSTR;

    // Build h1 stripe: rows k=0..15, lane covers 8 contiguous cols.
    {
        const float* a1p = A1 + ((size_t)((b * 5 + m) * 16 + j)) * 256 + lane * 8;
        v4f a1lo = *(const v4f*)(a1p);
        v4f a1hi = *(const v4f*)(a1p + 4);
        for (int k = 0; k < 16; ++k) {
            const float* a2p = A2 + ((size_t)((b * 5 + n) * 16 + k)) * 256 + lane * 8;
            v4f a2lo = *(const v4f*)(a2p);
            v4f a2hi = *(const v4f*)(a2p + 4);
            unsigned int pk[4];
            pk[0] = pack2bf(relu(a1lo[0] + a2lo[0]), relu(a1lo[1] + a2lo[1]));
            pk[1] = pack2bf(relu(a1lo[2] + a2lo[2]), relu(a1lo[3] + a2lo[3]));
            pk[2] = pack2bf(relu(a1hi[0] + a2hi[0]), relu(a1hi[1] + a2hi[1]));
            pk[3] = pack2bf(relu(a1hi[2] + a2hi[2]), relu(a1hi[3] + a2hi[3]));
            *(v4u*)(src + k * LSTR + lane * 8) = *(const v4u*)pk;   // ds_store_b128
        }
    }
    __syncthreads();

    for (int layer = 0; layer < 2; ++layer) {
        const unsigned short* W = layer ? W3_bf : W2_bf;
        for (int chunk = 0; chunk < 2; ++chunk) {
            v8f zero = {};
            v8f acc[8];
#pragma unroll
            for (int t = 0; t < 8; ++t) acc[t] = zero;
            for (int ks = 0; ks < 8; ++ks) {
                const int kb = ks * 32;
                Frag a = load_afrag_bf(src, LSTR, ln, kb, g);   // ds_load_b128 x2
#pragma unroll
                for (int nt2 = 0; nt2 < 8; ++nt2) {
                    const int nt = chunk * 8 + nt2;
                    Frag bfr = load_bfrag(W, 256, nt * 16 + ln, kb, g);  // global b128 x2
                    acc[nt2] = __builtin_amdgcn_wmma_f32_16x16x32_bf16(
                        false, a.v, false, bfr.v, (short)0, acc[nt2], false, false);
                }
            }
            if (layer == 0) {
#pragma unroll
                for (int nt2 = 0; nt2 < 8; ++nt2) {
                    const int nt = chunk * 8 + nt2;
#pragma unroll
                    for (int r = 0; r < 8; ++r) {
                        int M = g * 8 + r;
                        dst[(size_t)M * LSTR + nt * 16 + ln] = f2bf(relu(acc[nt2][r]));
                    }
                }
            } else {
#pragma unroll
                for (int nt2 = 0; nt2 < 8; ++nt2) {
                    const int nt = chunk * 8 + nt2;
                    float s = 0.0f;
#pragma unroll
                    for (int r = 0; r < 8; ++r) s += relu(acc[nt2][r]);
                    s += __shfl_xor(s, 16, 32);   // fold M=8..15 half onto same N
                    if (g == 0)
                        atomicAdd(&stats[((size_t)(b * 5 + m)) * 256 + nt * 16 + ln], s);
                }
            }
        }
        { unsigned short* t = src; src = dst; dst = t; }
        __syncthreads();
    }
}

// ------------------------------------------------------------ finalize ------
__global__ void k_final(const float* __restrict__ stats, const float* __restrict__ z,
                        float* __restrict__ outp, float* __restrict__ kl_out) {
    __shared__ float red[256];
    const int idx = blockIdx.x * 256 + threadIdx.x;   // 0..40959
    const int hh = idx & 127;
    const int bi = idx >> 7;                          // b*5 + i
    const float inv = 1.0f / 1280.0f;
    float mean = stats[(size_t)bi * 256 + hh] * inv;
    float var  = stats[(size_t)bi * 256 + 128 + hh] * inv;
    float lat = mean + var * z[idx];
    outp[idx] = lat;
    float ve = var + EPS_F;
    float d1 = (lat - mean) / ve;
    float lp1 = -0.5f * d1 * d1 - (__logf(ve) + 0.5f * LOG_2PI_F);
    float o1 = 1.0f + EPS_F;
    float d0 = lat / o1;
    float lp0 = -0.5f * d0 * d0 - (__logf(o1) + 0.5f * LOG_2PI_F);
    red[threadIdx.x] = (lp1 - lp0) * (1.0f / 40960.0f);
    __syncthreads();
    for (int s = 128; s > 0; s >>= 1) {
        if (threadIdx.x < s) red[threadIdx.x] += red[threadIdx.x + s];
        __syncthreads();
    }
    if (threadIdx.x == 0) atomicAdd(kl_out, red[0]);
}

// -------------------------------------------------------------- launch ------
extern "C" void kernel_launch(void* const* d_in, const int* in_sizes, int n_in,
                              void* d_out, int out_size, void* d_ws, size_t ws_size,
                              hipStream_t stream) {
    const float* x  = (const float*)d_in[0];   // (64,5,16,640)
    const float* z  = (const float*)d_in[1];   // (64,5,128)
    const float* We = (const float*)d_in[2];   // (128,640)
    const float* W1 = (const float*)d_in[3];   // (256,256)
    const float* W2 = (const float*)d_in[4];
    const float* W3 = (const float*)d_in[5];
    (void)in_sizes; (void)n_in; (void)out_size; (void)ws_size;

    char* ws = (char*)d_ws;
    unsigned short* We_bf  = (unsigned short*)(ws + 0);         // 163840 B
    unsigned short* W1_bf  = (unsigned short*)(ws + 163840);    // 131072 B
    unsigned short* W2_bf  = (unsigned short*)(ws + 294912);    // 131072 B
    unsigned short* W3_bf  = (unsigned short*)(ws + 425984);    // 131072 B
    unsigned short* out_bf = (unsigned short*)(ws + 557056);    // 1310720 B
    float* A1    = (float*)(ws + 1867776);                      // 5242880 B
    float* A2    = (float*)(ws + 7110656);                      // 5242880 B
    float* stats = (float*)(ws + 12353536);                     // 327680 B (end ~12.7 MB)

    float* outp = (float*)d_out;
    float* kl   = outp + 40960;

    k_init<<<256, 256, 0, stream>>>(We, W1, W2, W3, We_bf, W1_bf, W2_bf, W3_bf, stats, kl);
    k_encoder<<<320, 256, 0, stream>>>(x, We_bf, out_bf);
    k_layer1<<<dim3(320, 2), 512, 0, stream>>>(out_bf, W1_bf, A1, A2);
    size_t ldsb = 2u * 16u * 16u * LSTR * sizeof(unsigned short); // 270336 B
    k_pairmlp<<<1600, 512, ldsb, stream>>>(A1, A2, W2_bf, W3_bf, stats);
    k_final<<<160, 256, 0, stream>>>(stats, z, outp, kl);
}
